// PVConv_73718818669263
// MI455X (gfx1250) — matrix operand order repaired and checked by reading, not compiled, not run on hardware
//
#include <hip/hip_runtime.h>
#include <hip/hip_bf16.h>

typedef _Float16 half_t;
typedef __attribute__((ext_vector_type(16))) _Float16 v16h;
typedef __attribute__((ext_vector_type(8)))  _Float16 v8h;
typedef __attribute__((ext_vector_type(8)))  float    v8f;
typedef __attribute__((ext_vector_type(4)))  unsigned uint4v;
typedef __attribute__((ext_vector_type(8)))  int      int8v;
typedef __attribute__((ext_vector_type(4)))  int      int4v;

#define BB 8
#define CC 64
#define NN 65536
#define RR 32
#define RP 34
#define R3 (RR*RR*RR)
#define RP3 (RP*RP*RP)
// channel-last padded activation strides (in halfs)
#define SZC (CC)              // z step   = 64
#define SYC (RP*CC)           // y step   = 2176
#define SXC (RP*RP*CC)        // x step   = 73984
#define SBC (RP*RP*RP*CC)     // b step

#if defined(__has_builtin)
# if __has_builtin(__builtin_amdgcn_tensor_load_to_lds)
#  define USE_TDM 1
# endif
#endif

// ------------------------------------------------------------------
// 1) per-batch coord sum (for mean)
// ------------------------------------------------------------------
__global__ void k_mean(const float* __restrict__ coords, float* __restrict__ meansum) {
    __shared__ float red[256];
    int b = blockIdx.y;
    int n = blockIdx.x * 256 + threadIdx.x;
    float v[3];
    for (int d = 0; d < 3; ++d) v[d] = coords[(b * 3 + d) * NN + n];
    for (int d = 0; d < 3; ++d) {
        red[threadIdx.x] = v[d];
        __syncthreads();
        for (int s = 128; s > 0; s >>= 1) {
            if (threadIdx.x < s) red[threadIdx.x] += red[threadIdx.x + s];
            __syncthreads();
        }
        if (threadIdx.x == 0) atomicAdd(&meansum[b * 4 + d], red[0]);
        __syncthreads();
    }
}

// ------------------------------------------------------------------
// 2) per-batch max ||c - mean||  (bitwise atomicMax: norms >= 0)
// ------------------------------------------------------------------
__global__ void k_scale(const float* __restrict__ coords, const float* __restrict__ meansum,
                        unsigned* __restrict__ scaleu) {
    __shared__ float red[256];
    int b = blockIdx.y;
    int n = blockIdx.x * 256 + threadIdx.x;
    float s = 0.f;
    for (int d = 0; d < 3; ++d) {
        float c = coords[(b * 3 + d) * NN + n] - meansum[b * 4 + d] * (1.0f / NN);
        s += c * c;
    }
    s = sqrtf(s);
    red[threadIdx.x] = s;
    __syncthreads();
    for (int st = 128; st > 0; st >>= 1) {
        if (threadIdx.x < st) red[threadIdx.x] = fmaxf(red[threadIdx.x], red[threadIdx.x + st]);
        __syncthreads();
    }
    if (threadIdx.x == 0) atomicMax(&scaleu[b], __float_as_uint(red[0]));
}

// ------------------------------------------------------------------
// 3) normalize coords, scatter-add features + counts into voxel grid
// ------------------------------------------------------------------
__global__ void k_scatter(const float* __restrict__ coords, const float* __restrict__ feats,
                          const float* __restrict__ meansum, const unsigned* __restrict__ scaleu,
                          float* __restrict__ ncarr, float* __restrict__ voxsum,
                          float* __restrict__ cnt) {
    int t = blockIdx.x * 256 + threadIdx.x;           // over B*N
    int b = t / NN, n = t - b * NN;
    float inv = 1.0f / (__uint_as_float(scaleu[b]) * 2.0f);
    int id[3];
    for (int d = 0; d < 3; ++d) {
        float c = (coords[(b * 3 + d) * NN + n] - meansum[b * 4 + d] * (1.0f / NN)) * inv + 0.5f;
        float ncv = fminf(fmaxf(c * (float)RR, 0.0f), (float)(RR - 1));
        ncarr[(b * 3 + d) * NN + n] = ncv;
        id[d] = (int)rintf(ncv);                      // round-half-even like jnp.round
    }
    int flat = (id[0] * RR + id[1]) * RR + id[2];
    atomicAdd(&cnt[b * R3 + flat], 1.0f);
    const float* fb = feats + (size_t)b * CC * NN + n;
    float* vb = voxsum + (size_t)b * CC * R3 + flat;
    for (int ci = 0; ci < CC; ++ci) atomicAdd(vb + ci * R3, fb[(size_t)ci * NN]);
}

// ------------------------------------------------------------------
// 4) average & convert to f16 into zero-padded channel-last
//    voxpad[B][34][34][34][64]
// ------------------------------------------------------------------
__global__ void k_avg(const float* __restrict__ voxsum, const float* __restrict__ cnt,
                      half_t* __restrict__ voxpad) {
    size_t t = (size_t)blockIdx.x * 256 + threadIdx.x;    // B*R3*64
    int ci = (int)(t & 63);
    int vox = (int)((t >> 6) & (R3 - 1));
    int b = (int)(t >> 21);
    int z = vox & 31, y = (vox >> 5) & 31, x = vox >> 10;
    float v = voxsum[((size_t)(b * CC + ci)) * R3 + vox] / fmaxf(cnt[b * R3 + vox], 1.0f);
    voxpad[(size_t)b * SBC + (size_t)(x + 1) * SXC + (y + 1) * SYC + (z + 1) * SZC + ci] = (half_t)v;
}

// ------------------------------------------------------------------
// 5) fold BN (+ bias) into per-channel alpha/beta
//    bnp layout: a1[64] be1[64] a2[64] be2[64] a3[64] be3[64]
// ------------------------------------------------------------------
__global__ void k_bnprep(const float* cb1, const float* g1, const float* b1, const float* m1, const float* v1,
                         const float* cb2, const float* g2, const float* b2, const float* m2, const float* v2,
                         const float* mb3, const float* g3, const float* b3, const float* m3, const float* v3,
                         float* __restrict__ bnp) {
    int c = threadIdx.x;
    if (c >= CC) return;
    float a;
    a = g1[c] * rsqrtf(v1[c] + 1e-5f); bnp[c]       = a; bnp[64  + c] = (cb1[c] - m1[c]) * a + b1[c];
    a = g2[c] * rsqrtf(v2[c] + 1e-5f); bnp[128 + c] = a; bnp[192 + c] = (cb2[c] - m2[c]) * a + b2[c];
    a = g3[c] * rsqrtf(v3[c] + 1e-5f); bnp[256 + c] = a; bnp[320 + c] = (mb3[c] - m3[c]) * a + b3[c];
}

// ------------------------------------------------------------------
// 6) pack weights into the A-matrix (16x32 f16) lane layout, contiguous k = 16*(lane/16) + e
// ------------------------------------------------------------------
__global__ void k_pack(const float* __restrict__ w, half_t* __restrict__ dst, int ntap) {
    int t = blockIdx.x * 256 + threadIdx.x;
    int total = 4 * ntap * 2 * 512;
    if (t >= total) return;
    int e = t & 15, lane = (t >> 4) & 31, kc = (t >> 9) & 1;
    int tap = (t >> 10) % ntap, mblk = t / (1024 * ntap);
    int m = lane & 15, half = lane >> 4;
    int j = e >> 1, p = e & 1;
    int k = e + 16 * half;
    int co = mblk * 16 + m, ci = kc * 32 + k;
    dst[t] = (half_t)w[(co * CC + ci) * ntap + tap];
}

// ------------------------------------------------------------------
// 7) conv3x3x3 + BN + SiLU, 27-tap implicit GEMM (WMMA f16->f32).
//    Channel-last padded input; slab[9][34*64] staged by the TDM
//    (TENSOR_LOAD_TO_LDS, 2-D tile: dim0 = 3*34*64 contiguous halfs,
//    dim1 = 3 x-planes, stride = 34*34*64) and waited with
//    s_wait_tensorcnt.
// ------------------------------------------------------------------
__global__ __launch_bounds__(256) void k_conv(const half_t* __restrict__ in_pad,
                                              const half_t* __restrict__ wpack,
                                              const float* __restrict__ alpha,
                                              const float* __restrict__ beta,
                                              half_t* __restrict__ out_pad,
                                              float* __restrict__ out_f32,
                                              int mode) {
    __shared__ __align__(128) half_t slab[9 * RP * CC];   // [dx*3+dy][z][ci], 39168 B
    int wg = blockIdx.x;
    int b = wg >> 10, xy = wg & 1023, x = xy >> 5, y = xy & 31;
    int tid = threadIdx.x;

    const half_t* gsrc = in_pad + (size_t)b * SBC + (size_t)x * SXC + (size_t)y * SYC;

#ifdef USE_TDM
    if (tid < 32) {
        // Tensor DMA descriptor (ISA 8.3/8.4): 2-D tile, data_size=2B.
        unsigned long long ga = (unsigned long long)(uintptr_t)gsrc;
        unsigned ldsa = (unsigned)(uintptr_t)(void*)slab;   // low 32 bits = LDS offset
        uint4v g0;
        g0[0] = 1u;                                          // count=1, user mode
        g0[1] = ldsa;                                        // lds_addr
        g0[2] = (unsigned)(ga & 0xffffffffu);                // global_addr[31:0]
        g0[3] = (unsigned)((ga >> 32) & 0x01ffffffu) | (2u << 30);  // addr[56:32] | type=2
        int8v g1;
        g1[0] = 0x00010000;                 // data_size=1 -> 2 bytes
        g1[1] = (int)(6528u << 16);         // tensor_dim0[15:0] in bits 63:48
        g1[2] = (int)(3u << 16);            // tensor_dim1 in bits 95:80
        g1[3] = (int)(6528u << 16);         // tile_dim0 in bits 127:112  (3*34*64)
        g1[4] = 3;                          // tile_dim1 = 3 x-planes
        g1[5] = SXC;                        // tensor_dim0_stride = 34*34*64 elements
        g1[6] = 0;
        g1[7] = 0;
        int4v z4 = {0, 0, 0, 0};
        int8v z8 = {0, 0, 0, 0, 0, 0, 0, 0};
        __builtin_amdgcn_tensor_load_to_lds(g0, g1, z4, z4, z8, 0);
        __builtin_amdgcn_s_wait_tensorcnt(0);
    }
#else
    {   // fallback: layouts match, plain strided copy (u32 granularity)
        const unsigned* gs = (const unsigned*)gsrc;
        unsigned* sd = (unsigned*)slab;
        for (int i = tid; i < 9792; i += 256) {             // 3 blocks of 3264 u32
            int dx = i / 3264, rem = i - dx * 3264;
            sd[i] = gs[dx * (SXC / 2) + rem];
        }
    }
#endif
    __syncthreads();

    int wave = tid >> 5, lane = tid & 31;
    int mblk = wave & 3, nseg = wave >> 2;
    int nl = lane & 15, half = lane >> 4;

    v8f acc = {};
    const half_t* wbase = wpack + (size_t)mblk * 27 * 1024 + lane * 16;
    for (int tap = 0; tap < 27; ++tap) {
        int dx = tap / 9, dy = (tap / 3) % 3, dz = tap % 3;
        int zp = nseg * 16 + nl + dz;                       // padded z in [0,33]
        const half_t* brow = slab + (dx * 3 + dy) * (RP * CC) + zp * CC + 16 * half;
        if (tap < 26) __builtin_prefetch(wbase + (tap + 1) * 1024, 0, 0);
        v16h a0 = *(const v16h*)(wbase + tap * 1024);
        v16h b0 = *(const v16h*)(brow);
        acc = __builtin_amdgcn_wmma_f32_16x16x32_f16(false, a0, false, b0, (short)0, acc, false, false);
        v16h a1 = *(const v16h*)(wbase + tap * 1024 + 512);
        v16h b1 = *(const v16h*)(brow + 32);
        acc = __builtin_amdgcn_wmma_f32_16x16x32_f16(false, a1, false, b1, (short)0, acc, false, false);
    }

    int z = nseg * 16 + nl;
    int cobase = mblk * 16 + 8 * half;                      // C/D layout: M = r + 8*(lane/16)
    if (mode == 0) {
        v8h oh;
        for (int r = 0; r < 8; ++r) {
            int co = cobase + r;
            float v = acc[r] * alpha[co] + beta[co];
            v = v * (1.0f / (1.0f + __expf(-v)));           // SiLU
            oh[r] = (half_t)v;
        }
        *(v8h*)(out_pad + (size_t)b * SBC + (size_t)(x + 1) * SXC + (y + 1) * SYC +
                (z + 1) * SZC + cobase) = oh;
    } else {
        v8f of;
        for (int r = 0; r < 8; ++r) {
            int co = cobase + r;
            float v = acc[r] * alpha[co] + beta[co];
            v = v * (1.0f / (1.0f + __expf(-v)));           // SiLU
            of[r] = v;
        }
        *(v8f*)(out_f32 + ((size_t)b * R3 + ((x * RR + y) * RR + z)) * CC + cobase) = of;
    }
}

// ------------------------------------------------------------------
// 8) fused point MLP (WMMA) + BN + ReLU + trilinear devox + add.
//    vox2 is channel-last [B][32^3][64] -> 8 aligned v8f gathers/lane.
// ------------------------------------------------------------------
__global__ __launch_bounds__(256) void k_fused(const float* __restrict__ feats,
                                               const half_t* __restrict__ wmlp,
                                               const float* __restrict__ bnp,
                                               const float* __restrict__ ncarr,
                                               const float* __restrict__ vox2,
                                               float* __restrict__ out) {
    __shared__ __align__(32) half_t fld[32][CC];
    int wg = blockIdx.x;                                    // B * N/32
    int b = wg >> 11, p0 = (wg & 2047) * 32;
    int tid = threadIdx.x;
    for (int i = tid; i < 32 * CC; i += 256) {
        int pl = i & 31, ci = i >> 5;
        fld[pl][ci] = (half_t)feats[((size_t)(b * CC + ci)) * NN + p0 + pl];
    }
    __syncthreads();

    int wave = tid >> 5, lane = tid & 31;
    int mblk = wave & 3, pseg = wave >> 2;
    int nl = lane & 15, half = lane >> 4;

    v8f acc = {};
    const half_t* wb = wmlp + mblk * 1024 + lane * 16;
    const half_t* br = &fld[pseg * 16 + nl][16 * half];
    v16h a0 = *(const v16h*)wb;
    v16h b0 = *(const v16h*)br;
    acc = __builtin_amdgcn_wmma_f32_16x16x32_f16(false, a0, false, b0, (short)0, acc, false, false);
    v16h a1 = *(const v16h*)(wb + 512);
    v16h b1 = *(const v16h*)(br + 32);
    acc = __builtin_amdgcn_wmma_f32_16x16x32_f16(false, a1, false, b1, (short)0, acc, false, false);

    int p = p0 + pseg * 16 + nl;
    float fx = ncarr[(b * 3 + 0) * NN + p];
    float fy = ncarr[(b * 3 + 1) * NN + p];
    float fz = ncarr[(b * 3 + 2) * NN + p];
    float lx = floorf(fx), ly = floorf(fy), lz = floorf(fz);
    float gx = fx - lx, gy = fy - ly, gz = fz - lz;
    int x0 = (int)lx, y0 = (int)ly, z0 = (int)lz;
    int x1 = min(x0 + 1, RR - 1), y1 = min(y0 + 1, RR - 1), z1 = min(z0 + 1, RR - 1);
    int o000 = (x0 * RR + y0) * RR + z0, o001 = (x0 * RR + y0) * RR + z1;
    int o010 = (x0 * RR + y1) * RR + z0, o011 = (x0 * RR + y1) * RR + z1;
    int o100 = (x1 * RR + y0) * RR + z0, o101 = (x1 * RR + y0) * RR + z1;
    int o110 = (x1 * RR + y1) * RR + z0, o111 = (x1 * RR + y1) * RR + z1;

    int cobase = mblk * 16 + 8 * half;
    const float* vb = vox2 + (size_t)b * R3 * CC + cobase;
    v8f dv;
    dv  = *(const v8f*)(vb + (size_t)o000 * CC) * ((1 - gx) * (1 - gy) * (1 - gz));
    dv += *(const v8f*)(vb + (size_t)o001 * CC) * ((1 - gx) * (1 - gy) * gz);
    dv += *(const v8f*)(vb + (size_t)o010 * CC) * ((1 - gx) * gy * (1 - gz));
    dv += *(const v8f*)(vb + (size_t)o011 * CC) * ((1 - gx) * gy * gz);
    dv += *(const v8f*)(vb + (size_t)o100 * CC) * (gx * (1 - gy) * (1 - gz));
    dv += *(const v8f*)(vb + (size_t)o101 * CC) * (gx * (1 - gy) * gz);
    dv += *(const v8f*)(vb + (size_t)o110 * CC) * (gx * gy * (1 - gz));
    dv += *(const v8f*)(vb + (size_t)o111 * CC) * (gx * gy * gz);

    for (int r = 0; r < 8; ++r) {
        int co = cobase + r;
        float ptv = fmaxf(acc[r] * bnp[256 + co] + bnp[320 + co], 0.0f);
        out[((size_t)(b * CC + co)) * NN + p] = ptv + dv[r];
    }
}

// ------------------------------------------------------------------
extern "C" void kernel_launch(void* const* d_in, const int* in_sizes, int n_in,
                              void* d_out, int out_size, void* d_ws, size_t ws_size,
                              hipStream_t stream) {
    const float* features = (const float*)d_in[0];
    const float* coords   = (const float*)d_in[1];
    const float* time_emb = (const float*)d_in[2];
    const float* conv1_w  = (const float*)d_in[3];
    const float* conv1_b  = (const float*)d_in[4];
    const float* bn1_g = (const float*)d_in[5],  *bn1_b = (const float*)d_in[6];
    const float* bn1_m = (const float*)d_in[7],  *bn1_v = (const float*)d_in[8];
    const float* conv2_w  = (const float*)d_in[9];
    const float* conv2_b  = (const float*)d_in[10];
    const float* bn2_g = (const float*)d_in[11], *bn2_b = (const float*)d_in[12];
    const float* bn2_m = (const float*)d_in[13], *bn2_v = (const float*)d_in[14];
    const float* mlp_w    = (const float*)d_in[15];
    const float* mlp_b    = (const float*)d_in[16];
    const float* bn3_g = (const float*)d_in[17], *bn3_b = (const float*)d_in[18];
    const float* bn3_m = (const float*)d_in[19], *bn3_v = (const float*)d_in[20];

    char* ws = (char*)d_ws;
    size_t off = 0;
    auto take = [&](size_t bytes) { size_t o = off; off = (off + bytes + 255) & ~(size_t)255; return o; };
    float*    meansum = (float*)   (ws + take(BB * 4 * sizeof(float)));
    unsigned* scaleu  = (unsigned*)(ws + take(BB * sizeof(unsigned)));
    float*    bnp     = (float*)   (ws + take(6 * CC * sizeof(float)));
    half_t*   wpack1  = (half_t*)  (ws + take((size_t)4 * 27 * 1024 * sizeof(half_t)));
    half_t*   wpack2  = (half_t*)  (ws + take((size_t)4 * 27 * 1024 * sizeof(half_t)));
    half_t*   wmlpp   = (half_t*)  (ws + take((size_t)4 * 1024 * sizeof(half_t)));
    float*    cnt     = (float*)   (ws + take((size_t)BB * R3 * sizeof(float)));
    float*    ncarr   = (float*)   (ws + take((size_t)BB * 3 * NN * sizeof(float)));
    float*    voxsum  = (float*)   (ws + take((size_t)BB * CC * R3 * sizeof(float)));   // reused as conv2 out
    half_t*   voxpad  = (half_t*)  (ws + take((size_t)BB * CC * RP3 * sizeof(half_t)));
    half_t*   actpad  = (half_t*)  (ws + take((size_t)BB * CC * RP3 * sizeof(half_t)));

    (void)hipMemsetAsync(meansum, 0, BB * 4 * sizeof(float), stream);
    (void)hipMemsetAsync(scaleu, 0, BB * sizeof(unsigned), stream);
    (void)hipMemsetAsync(cnt, 0, (size_t)BB * R3 * sizeof(float), stream);
    (void)hipMemsetAsync(voxsum, 0, (size_t)BB * CC * R3 * sizeof(float), stream);
    (void)hipMemsetAsync(voxpad, 0, (size_t)BB * CC * RP3 * sizeof(half_t), stream);
    (void)hipMemsetAsync(actpad, 0, (size_t)BB * CC * RP3 * sizeof(half_t), stream);

    k_bnprep<<<1, 64, 0, stream>>>(conv1_b, bn1_g, bn1_b, bn1_m, bn1_v,
                                   conv2_b, bn2_g, bn2_b, bn2_m, bn2_v,
                                   mlp_b,   bn3_g, bn3_b, bn3_m, bn3_v, bnp);
    k_pack<<<(4 * 27 * 1024 + 255) / 256, 256, 0, stream>>>(conv1_w, wpack1, 27);
    k_pack<<<(4 * 27 * 1024 + 255) / 256, 256, 0, stream>>>(conv2_w, wpack2, 27);
    k_pack<<<(4 * 1024 + 255) / 256, 256, 0, stream>>>(mlp_w, wmlpp, 1);

    k_mean<<<dim3(NN / 256, BB), 256, 0, stream>>>(coords, meansum);
    k_scale<<<dim3(NN / 256, BB), 256, 0, stream>>>(coords, meansum, scaleu);
    k_scatter<<<(BB * NN) / 256, 256, 0, stream>>>(coords, features, meansum, scaleu,
                                                   ncarr, voxsum, cnt);
    k_avg<<<(BB * CC * R3) / 256, 256, 0, stream>>>(voxsum, cnt, voxpad);

    k_conv<<<BB * 1024, 256, 0, stream>>>(voxpad, wpack1, bnp + 0,   bnp + 64,  actpad, nullptr, 0);
    k_conv<<<BB * 1024, 256, 0, stream>>>(actpad, wpack2, bnp + 128, bnp + 192, nullptr, voxsum, 1);

    float* fused = (float*)d_out;
    k_fused<<<BB * (NN / 32), 256, 0, stream>>>(features, wmlpp, bnp, ncarr, voxsum, fused);

    (void)hipMemcpyAsync(fused + (size_t)BB * CC * NN, coords,
                         (size_t)BB * 3 * NN * sizeof(float), hipMemcpyDeviceToDevice, stream);
    (void)hipMemcpyAsync(fused + (size_t)BB * CC * NN + (size_t)BB * 3 * NN, time_emb,
                         (size_t)BB * CC * sizeof(float), hipMemcpyDeviceToDevice, stream);
    (void)in_sizes; (void)n_in; (void)out_size; (void)ws_size;
}
